// GILRLayer_22986664968458
// MI455X (gfx1250) — compile-verified, matrix operand order checked by reading, not run on hardware
//
#include <hip/hip_runtime.h>
#include <hip/hip_bf16.h>
#include <stdint.h>

#define D_MODEL 1024
#define BATCH   4
#define SEQ     4096
#define M_ROWS  (BATCH * SEQ)      // 16384
#define N1      (3 * D_MODEL)      // 3072
#define NCHUNK  64
#define CHUNK   (SEQ / NCHUNK)     // 64
#define LN_EPS  1e-5f

typedef __attribute__((ext_vector_type(16))) __bf16 v16bf;
typedef __attribute__((ext_vector_type(8)))  float  v8f;
typedef __attribute__((ext_vector_type(4)))  int    v4i_;

// ---------------- gfx1250 async global->LDS (guarded) ----------------

#if defined(__AMDGCN__) && __has_builtin(__builtin_amdgcn_global_load_async_to_lds_b128)
#define USE_ASYNC_LDS 1
typedef __attribute__((address_space(1))) v4i_* g_b128_ptr;
typedef __attribute__((address_space(3))) v4i_* l_b128_ptr;
#else
#define USE_ASYNC_LDS 0
#endif

__device__ __forceinline__ void copy16_g2l(unsigned short* lds_dst,
                                           const unsigned short* gsrc) {
#if USE_ASYNC_LDS
  __builtin_amdgcn_global_load_async_to_lds_b128(
      (g_b128_ptr)(v4i_*)(void*)gsrc,
      (l_b128_ptr)(v4i_*)(void*)lds_dst,
      /*offset=*/0, /*cpol=*/0);
#else
  *(uint4*)lds_dst = *(const uint4*)gsrc;
#endif
}

template <int N>
__device__ __forceinline__ void wait_async() {
#if USE_ASYNC_LDS
#if __has_builtin(__builtin_amdgcn_s_wait_asynccnt)
  __builtin_amdgcn_s_wait_asynccnt(N);
#else
  asm volatile("s_wait_asynccnt %0" ::"n"(N) : "memory");
#endif
#endif
}

// ---------------- helpers ----------------

__device__ __forceinline__ unsigned short f2bf(float x) {
  union { float f; unsigned int u; } c; c.f = x;
  unsigned int u = c.u;
  if ((u & 0x7F800000u) != 0x7F800000u)     // normal/denorm: round-to-nearest-even
    u += 0x7FFFu + ((u >> 16) & 1u);
  return (unsigned short)(u >> 16);
}

__device__ __forceinline__ float sigmoidf_(float x) {
  return 1.0f / (1.0f + __expf(-x));
}

// ---------------- conversion kernels ----------------

__global__ void cvt_bf16_kernel(const float* __restrict__ in,
                                unsigned short* __restrict__ out, int n) {
  int i = blockIdx.x * blockDim.x + threadIdx.x;
  if (i < n) out[i] = f2bf(in[i]);
}

// W [K, N] fp32 (row-major) -> Wt [N, K] bf16 (row-major)
__global__ void cvt_transpose_kernel(const float* __restrict__ W,
                                     unsigned short* __restrict__ Wt,
                                     int K, int N) {
  long long i = (long long)blockIdx.x * blockDim.x + threadIdx.x;
  if (i >= (long long)K * N) return;
  int k = (int)(i % K);
  int n = (int)(i / K);
  Wt[(long long)n * K + k] = f2bf(W[(long long)k * N + n]);
}

// ---------------- WMMA GEMM: C[M,N] = A[M,K](bf16) * Bt[N,K](bf16)^T + bias ----
// Block: 256 threads = 8 waves. Block tile 128x64; wave tile 32x32 (4 WMMAs).
// K is consumed in 32-wide slabs, double-buffered through LDS via async copies.

__global__ void __launch_bounds__(256) wmma_gemm_bias(
    const unsigned short* __restrict__ A,   // [M,K] bf16 row-major
    const unsigned short* __restrict__ Bt,  // [N,K] bf16 row-major (B transposed)
    const float* __restrict__ bias,         // [N]
    float* __restrict__ C,                  // [M,N] fp32
    int M, int N, int K)
{
  __shared__ __align__(16) unsigned short lA[2][128 * 32];  // 16 KB
  __shared__ __align__(16) unsigned short lB[2][64 * 32];   //  8 KB

  const int tid  = threadIdx.x;
  const int lane = tid & 31;
  const int wave = tid >> 5;
  const int wm   = wave & 3;                 // 32-row block within 128
  const int wn   = wave >> 2;                // 32-col block within 64
  const int nbn  = N >> 6;
  const int bm   = (int)blockIdx.x / nbn;
  const int bn   = (int)blockIdx.x % nbn;

  const int half = lane >> 4;                // K-pattern half select
  const int l15  = lane & 15;

  // --- cooperative copy maps (16B chunks) ---
  // A slab: 128 rows x 32 k = 8KB = 512 chunks -> 2 per thread
  const int ac0 = tid, ac1 = tid + 256;
  const unsigned short* gA0 = A + (long long)(bm * 128 + (ac0 >> 2)) * K + (ac0 & 3) * 8;
  const unsigned short* gA1 = A + (long long)(bm * 128 + (ac1 >> 2)) * K + (ac1 & 3) * 8;
  const int sa0 = (ac0 >> 2) * 32 + (ac0 & 3) * 8;
  const int sa1 = (ac1 >> 2) * 32 + (ac1 & 3) * 8;
  // B slab: 64 rows x 32 k = 4KB = 256 chunks -> 1 per thread
  const unsigned short* gB0 = Bt + (long long)(bn * 64 + (tid >> 2)) * K + (tid & 3) * 8;
  const int sb0 = (tid >> 2) * 32 + (tid & 3) * 8;

  v8f acc[2][2] = {};
  const int nk = K >> 5;

  // prologue: slab 0 into buffer 0
  copy16_g2l(&lA[0][sa0], gA0);
  copy16_g2l(&lA[0][sa1], gA1);
  copy16_g2l(&lB[0][sb0], gB0);

  for (int ks = 0; ks < nk; ++ks) {
    const int buf = ks & 1;
    if (ks + 1 < nk) {
      const int k0 = (ks + 1) << 5;
      copy16_g2l(&lA[buf ^ 1][sa0], gA0 + k0);
      copy16_g2l(&lA[buf ^ 1][sa1], gA1 + k0);
      copy16_g2l(&lB[buf ^ 1][sb0], gB0 + k0);
      wait_async<3>();                       // current slab's 3 copies done
    } else {
      wait_async<0>();
    }
    __syncthreads();

    // --- load fragments from LDS (ISA 16-bit A/B layout) and multiply ---
    union Frag { v16bf v; uint4 q[2]; } fa[2], fb[2];
#pragma unroll
    for (int s = 0; s < 2; ++s) {
      const char* pa = (const char*)&lA[buf][(wm * 32 + s * 16 + l15) * 32] + (half << 4);
      fa[s].q[0] = *(const uint4*)pa;        // K = kb .. kb+7
      fa[s].q[1] = *(const uint4*)(pa + 32); // K = 16+kb .. 16+kb+7
      const char* pb = (const char*)&lB[buf][(wn * 32 + s * 16 + l15) * 32] + (half << 4);
      fb[s].q[0] = *(const uint4*)pb;
      fb[s].q[1] = *(const uint4*)(pb + 32);
    }
#pragma unroll
    for (int sm = 0; sm < 2; ++sm)
#pragma unroll
      for (int sn = 0; sn < 2; ++sn)
        acc[sm][sn] = __builtin_amdgcn_wmma_f32_16x16x32_bf16(
            false, fa[sm].v, false, fb[sn].v, (short)0, acc[sm][sn], false, false);

    __syncthreads();                         // protect buf before it is refilled
  }

  // --- epilogue: C/D layout (lane half -> rows +8), bias fused ---
#pragma unroll
  for (int sn = 0; sn < 2; ++sn) {
    const int col = bn * 64 + wn * 32 + sn * 16 + l15;
    const float bv = bias[col];
#pragma unroll
    for (int sm = 0; sm < 2; ++sm) {
      const int row0 = bm * 128 + wm * 32 + sm * 16 + (half << 3);
#pragma unroll
      for (int i = 0; i < 8; ++i)
        C[(long long)(row0 + i) * N + col] = acc[sm][sn][i] + bv;
    }
  }
}

// ---------------- chunked scan: h_t = a_t*h_{t-1} + (1-a_t)*v_t, a=sigmoid(f) ----

__global__ void scan_carry_kernel(const float* __restrict__ u,
                                  float* __restrict__ cA,
                                  float* __restrict__ cB) {
  int tid = blockIdx.x * blockDim.x + threadIdx.x;     // (b*NCHUNK + chunk)*D + ch
  if (tid >= BATCH * NCHUNK * D_MODEL) return;
  int ch    = tid % D_MODEL;
  int chunk = (tid / D_MODEL) % NCHUNK;
  int b     = tid / (D_MODEL * NCHUNK);
  float Ap = 1.0f, Bc = 0.0f;
  long long base = ((long long)b * SEQ + (long long)chunk * CHUNK) * N1;
  for (int t = 0; t < CHUNK; ++t) {
    long long row = base + (long long)t * N1;
    float f = u[row + 2 * D_MODEL + ch];
    float v = u[row + ch];
    float a = sigmoidf_(f);
    Ap *= a;
    Bc  = a * Bc + (1.0f - a) * v;
  }
  cA[tid] = Ap;
  cB[tid] = Bc;
}

__global__ void scan_combine_kernel(const float* __restrict__ cA,
                                    const float* __restrict__ cB,
                                    float* __restrict__ Hprev) {
  int tid = blockIdx.x * blockDim.x + threadIdx.x;     // b*D + ch
  if (tid >= BATCH * D_MODEL) return;
  int ch = tid % D_MODEL;
  int b  = tid / D_MODEL;
  float H = 0.0f;
  for (int c = 0; c < NCHUNK; ++c) {
    long long idx = ((long long)b * NCHUNK + c) * D_MODEL + ch;
    Hprev[idx] = H;
    H = cA[idx] * H + cB[idx];
  }
}

__global__ void scan_apply_kernel(const float* __restrict__ u,
                                  const float* __restrict__ Hprev,
                                  float* __restrict__ g) {
  int tid = blockIdx.x * blockDim.x + threadIdx.x;
  if (tid >= BATCH * NCHUNK * D_MODEL) return;
  int ch    = tid % D_MODEL;
  int chunk = (tid / D_MODEL) % NCHUNK;
  int b     = tid / (D_MODEL * NCHUNK);
  float h = Hprev[tid];
  long long base  = ((long long)b * SEQ + (long long)chunk * CHUNK) * N1;
  long long gbase = ((long long)b * SEQ + (long long)chunk * CHUNK) * D_MODEL;
  for (int t = 0; t < CHUNK; ++t) {
    long long row = base + (long long)t * N1;
    float f = u[row + 2 * D_MODEL + ch];
    float v = u[row + ch];
    float o = u[row + D_MODEL + ch];
    float a = sigmoidf_(f);
    h = a * h + (1.0f - a) * v;
    float so = o * sigmoidf_(o);
    g[gbase + (long long)t * D_MODEL + ch] = h * so;
  }
}

// ---------------- LayerNorm (per row of D_MODEL) -> bf16 ----------------

__global__ void __launch_bounds__(256) ln_bf16_kernel(
    const float* __restrict__ g,
    const float* __restrict__ gamma,
    const float* __restrict__ beta,
    unsigned short* __restrict__ ghat) {
  __shared__ float ssum[256];
  __shared__ float ssq[256];
  const int row = blockIdx.x;
  const float* gr = g + (long long)row * D_MODEL;
  float s = 0.0f, q = 0.0f;
  for (int i = threadIdx.x; i < D_MODEL; i += 256) {
    float x = gr[i]; s += x; q += x * x;
  }
  ssum[threadIdx.x] = s; ssq[threadIdx.x] = q;
  __syncthreads();
  for (int off = 128; off > 0; off >>= 1) {
    if (threadIdx.x < off) {
      ssum[threadIdx.x] += ssum[threadIdx.x + off];
      ssq[threadIdx.x]  += ssq[threadIdx.x + off];
    }
    __syncthreads();
  }
  float mu  = ssum[0] * (1.0f / D_MODEL);
  float var = ssq[0] * (1.0f / D_MODEL) - mu * mu;
  float inv = rsqrtf(var + LN_EPS);
  for (int i = threadIdx.x; i < D_MODEL; i += 256) {
    float y = (gr[i] - mu) * inv * gamma[i] + beta[i];
    ghat[(long long)row * D_MODEL + i] = f2bf(y);
  }
}

// ---------------- launch ----------------

extern "C" void kernel_launch(void* const* d_in, const int* in_sizes, int n_in,
                              void* d_out, int out_size, void* d_ws, size_t ws_size,
                              hipStream_t stream) {
  const float* x     = (const float*)d_in[0];
  const float* W_in  = (const float*)d_in[1];
  const float* b_in  = (const float*)d_in[2];
  const float* gamma = (const float*)d_in[3];
  const float* beta  = (const float*)d_in[4];
  const float* W_out = (const float*)d_in[5];
  const float* b_out = (const float*)d_in[6];

  char* ws = (char*)d_ws;
  size_t off = 0;
  auto wsalloc = [&](size_t bytes) -> void* {
    void* p = ws + off;
    off = (off + bytes + 255) & ~(size_t)255;
    return p;
  };

  unsigned short* x_bf  = (unsigned short*)wsalloc((size_t)M_ROWS * D_MODEL * 2);
  unsigned short* w1t   = (unsigned short*)wsalloc((size_t)N1 * D_MODEL * 2);
  unsigned short* w2t   = (unsigned short*)wsalloc((size_t)D_MODEL * D_MODEL * 2);
  float*          u     = (float*)wsalloc((size_t)M_ROWS * N1 * 4);
  float*          cA    = (float*)wsalloc((size_t)BATCH * NCHUNK * D_MODEL * 4);
  float*          cB    = (float*)wsalloc((size_t)BATCH * NCHUNK * D_MODEL * 4);
  float*          Hprev = (float*)wsalloc((size_t)BATCH * NCHUNK * D_MODEL * 4);
  float*          g     = (float*)wsalloc((size_t)M_ROWS * D_MODEL * 4);
  unsigned short* ghat  = (unsigned short*)wsalloc((size_t)M_ROWS * D_MODEL * 2);
  (void)ws_size; (void)in_sizes; (void)n_in; (void)out_size;

  // 1) precision conversions
  {
    int n = M_ROWS * D_MODEL;
    cvt_bf16_kernel<<<(n + 255) / 256, 256, 0, stream>>>(x, x_bf, n);
  }
  {
    long long total = (long long)D_MODEL * N1;
    cvt_transpose_kernel<<<(int)((total + 255) / 256), 256, 0, stream>>>(W_in, w1t, D_MODEL, N1);
  }
  {
    long long total = (long long)D_MODEL * D_MODEL;
    cvt_transpose_kernel<<<(int)((total + 255) / 256), 256, 0, stream>>>(W_out, w2t, D_MODEL, D_MODEL);
  }

  // 2) u = x @ W_in + b_in   (WMMA bf16, fp32 accumulate; async-LDS pipeline)
  {
    int blocks = (M_ROWS / 128) * (N1 / 64);        // 6144
    wmma_gemm_bias<<<blocks, 256, 0, stream>>>(x_bf, w1t, b_in, u,
                                               M_ROWS, N1, D_MODEL);
  }

  // 3) chunked associative scan
  {
    int n = BATCH * NCHUNK * D_MODEL;               // 262144
    scan_carry_kernel<<<n / 256, 256, 0, stream>>>(u, cA, cB);
    int m = BATCH * D_MODEL;                        // 4096
    scan_combine_kernel<<<m / 256, 256, 0, stream>>>(cA, cB, Hprev);
    scan_apply_kernel<<<n / 256, 256, 0, stream>>>(u, Hprev, g);
  }

  // 4) LayerNorm -> bf16
  ln_bf16_kernel<<<M_ROWS, 256, 0, stream>>>(g, gamma, beta, ghat);

  // 5) y = g_hat @ W_out + b_out
  {
    int blocks = (M_ROWS / 128) * (D_MODEL / 64);   // 2048
    wmma_gemm_bias<<<blocks, 256, 0, stream>>>(ghat, w2t, b_out, (float*)d_out,
                                               M_ROWS, D_MODEL, D_MODEL);
  }
}
